// MultiGCN_57690000720658
// MI455X (gfx1250) — matrix operand order repaired and checked by reading, not compiled, not run on hardware
//
#include <hip/hip_runtime.h>
#include <hip/hip_bf16.h>
#include <math.h>

// ---------------------------------------------------------------------------
// CDNA5 WMMA types
// ---------------------------------------------------------------------------
typedef __attribute__((ext_vector_type(16))) _Float16 v16h;
typedef __attribute__((ext_vector_type(8)))  _Float16 v8h;
typedef __attribute__((ext_vector_type(4)))  _Float16 v4h;
typedef __attribute__((ext_vector_type(8)))  float    v8f;

#define BLK_M 64
#define BLK_N 64
#define KSTEP 32
#define ROWH  40   // halves per LDS row (80 B: 16-B aligned, bank-spread)

// ---------------------------------------------------------------------------
// GEMM: C[M,N] = act(A[M,K] @ W[K,N] + bias[N])
// f32 in memory, f16 WMMA with f32 accumulate.
// Block = 256 threads = 8 waves = 4 (M) x 2 (N) wave grid; each wave owns a
// 16x32 strip of C -> TWO WMMAs per K-slab with the A fragment reused.
// Software-pipelined, double-buffered LDS; tiles stored in WMMA *fragment
// order* so each lane loads fragments as contiguous ds_load_b128.
//   A frag:  k in slab -> h=k>>4,     e=k&15; lane m = lane&15
//   B frag:  k in slab -> h=k>>4,     e=k&15;             lane n = lane&15
// ALIGNED: requires M%64==0, N%64==0, K%4==0 (true for all GCN/embed GEMMs).
// ---------------------------------------------------------------------------
template <bool RELU, bool ALIGNED>
__global__ __launch_bounds__(256)
void gemm_bias_kernel(const float* __restrict__ A, const float* __restrict__ W,
                      const float* __restrict__ bias, float* __restrict__ C,
                      int M, int N, int K)
{
    __shared__ _Float16 sAf[2][BLK_M][ROWH];   // [buf][row][h*16+e]
    __shared__ _Float16 sBf[2][BLK_N][ROWH];   // [buf][col][h*16+e]

    const int tid  = threadIdx.x;
    const int lane = tid & 31;
    const int wave = tid >> 5;      // 0..7
    const int wm   = wave >> 1;     // 0..3  (16-row tile)
    const int wn   = wave & 1;      // 0..1  (32-col group)
    const int m0   = blockIdx.y * BLK_M;
    const int n0   = blockIdx.x * BLK_N;
    const int mrow = lane & 15;
    const int hi   = lane >> 4;

    // persistent per-thread staging coordinates
    const int ar  = tid >> 3;             // A rows ar and ar+32   (0..31)
    const int ac4 = (tid & 7) << 2;       // A k offset            (0,4,...,28)
    const int bn  = tid & 63;             // B col in slab         (0..63)
    const int bk4 = (tid >> 6) << 2;      // B k group 0 (0,4,8,12); group 1 = +16

    // staged values (registers)
    float a0, a1, a2, a3, a4, a5, a6, a7;          // rows ar / ar+32
    float b0, b1, b2, b3, b4, b5, b6, b7;

    // ---- issue global loads for slab at k0 ----
    auto gload = [&](int k0) {
        if (ALIGNED && (k0 + KSTEP <= K)) {        // uniform branch: fast path
            const float4 f0 = *(const float4*)(A + (size_t)(m0 + ar) * K + (k0 + ac4));
            const float4 f1 = *(const float4*)(A + (size_t)(m0 + ar + 32) * K + (k0 + ac4));
            a0 = f0.x; a1 = f0.y; a2 = f0.z; a3 = f0.w;
            a4 = f1.x; a5 = f1.y; a6 = f1.z; a7 = f1.w;
            const float* wp = W + (size_t)(k0 + bk4) * N + (n0 + bn);
            b0 = wp[0];
            b1 = wp[(size_t)N];
            b2 = wp[(size_t)N * 2];
            b3 = wp[(size_t)N * 3];
            const float* wq = wp + (size_t)N * 16;
            b4 = wq[0];
            b5 = wq[(size_t)N];
            b6 = wq[(size_t)N * 2];
            b7 = wq[(size_t)N * 3];
        } else {                                    // branchless clamped path
            const int gm0 = (ALIGNED || m0 + ar < M)      ? (m0 + ar)      : (M - 1);
            const int gm1 = (ALIGNED || m0 + ar + 32 < M) ? (m0 + ar + 32) : (M - 1);
            float av[8];
            #pragma unroll
            for (int j = 0; j < 4; ++j) {
                int gk = k0 + ac4 + j;
                int ck = (gk < K) ? gk : (K - 1);
                float v0 = A[(size_t)gm0 * K + ck];
                float v1 = A[(size_t)gm1 * K + ck];
                av[j]     = (gk < K) ? v0 : 0.0f;
                av[4 + j] = (gk < K) ? v1 : 0.0f;
            }
            a0 = av[0]; a1 = av[1]; a2 = av[2]; a3 = av[3];
            a4 = av[4]; a5 = av[5]; a6 = av[6]; a7 = av[7];
            const int gn = (ALIGNED || n0 + bn < N) ? (n0 + bn) : (N - 1);
            float bv[8];
            #pragma unroll
            for (int g = 0; g < 2; ++g)
                #pragma unroll
                for (int j = 0; j < 4; ++j) {
                    int gk = k0 + bk4 + g * 16 + j;
                    int ck = (gk < K) ? gk : (K - 1);
                    float v = W[(size_t)ck * N + gn];
                    bv[g * 4 + j] = (gk < K) ? v : 0.0f;
                }
            b0 = bv[0]; b1 = bv[1]; b2 = bv[2]; b3 = bv[3];
            b4 = bv[4]; b5 = bv[5]; b6 = bv[6]; b7 = bv[7];
        }
    };

    // ---- convert + store staged values into fragment-ordered LDS ----
    auto lstore = [&](int buf) {
        const int he = ac4;
        v4h ha;
        ha[0] = (_Float16)a0; ha[1] = (_Float16)a1;
        ha[2] = (_Float16)a2; ha[3] = (_Float16)a3;
        *(v4h*)&sAf[buf][ar][he] = ha;             // ds_store_b64
        ha[0] = (_Float16)a4; ha[1] = (_Float16)a5;
        ha[2] = (_Float16)a6; ha[3] = (_Float16)a7;
        *(v4h*)&sAf[buf][ar + 32][he] = ha;        // ds_store_b64
        v4h hb;
        hb[0] = (_Float16)b0; hb[1] = (_Float16)b1;
        hb[2] = (_Float16)b2; hb[3] = (_Float16)b3;
        *(v4h*)&sBf[buf][bn][bk4] = hb;            // h=0, e=bk4..bk4+3
        hb[0] = (_Float16)b4; hb[1] = (_Float16)b5;
        hb[2] = (_Float16)b6; hb[3] = (_Float16)b7;
        *(v4h*)&sBf[buf][bn][16 + bk4] = hb;       // h=1, e=bk4..bk4+3
    };

    v8f acc0 = {}, acc1 = {};
    const int nslab = (K + KSTEP - 1) / KSTEP;

    gload(0);
    int buf = 0;
    for (int s = 0; s < nslab; ++s) {
        lstore(buf);
        if (s + 1 < nslab) gload((s + 1) * KSTEP);   // overlap with compute below
        __syncthreads();

        // A fragment (reused by both WMMAs): two ds_load_b128
        const v8h alo = *(const v8h*)&sAf[buf][wm * 16 + mrow][hi * 16];
        const v8h ahi = *(const v8h*)&sAf[buf][wm * 16 + mrow][hi * 16 + 8];
        const v16h af = __builtin_shufflevector(alo, ahi, 0,1,2,3,4,5,6,7,8,9,10,11,12,13,14,15);
        // B fragments for the two 16-col tiles of this wave's 32-col group
        const int bc0 = wn * 32 + mrow;
        const v8h b0lo = *(const v8h*)&sBf[buf][bc0][hi * 16];
        const v8h b0hi = *(const v8h*)&sBf[buf][bc0][hi * 16 + 8];
        const v8h b1lo = *(const v8h*)&sBf[buf][bc0 + 16][hi * 16];
        const v8h b1hi = *(const v8h*)&sBf[buf][bc0 + 16][hi * 16 + 8];
        const v16h bf0 = __builtin_shufflevector(b0lo, b0hi, 0,1,2,3,4,5,6,7,8,9,10,11,12,13,14,15);
        const v16h bf1 = __builtin_shufflevector(b1lo, b1hi, 0,1,2,3,4,5,6,7,8,9,10,11,12,13,14,15);

        acc0 = __builtin_amdgcn_wmma_f32_16x16x32_f16(
                   false, af, false, bf0, (short)0, acc0, false, false);
        acc1 = __builtin_amdgcn_wmma_f32_16x16x32_f16(
                   false, af, false, bf1, (short)0, acc1, false, false);
        buf ^= 1;
    }

    // ---- epilogue: C/D layout VGPR r -> M = r + 8*laneHi ----
    const int mbase = m0 + wm * 16 + hi * 8;
    const int nc0   = n0 + wn * 32 + mrow;
    if (ALIGNED) {
        const float bv0 = bias[nc0];
        const float bv1 = bias[nc0 + 16];
        #pragma unroll
        for (int r = 0; r < 8; ++r) {
            float v0 = acc0[r] + bv0;
            float v1 = acc1[r] + bv1;
            if (RELU) { v0 = v0 > 0.0f ? v0 : 0.0f; v1 = v1 > 0.0f ? v1 : 0.0f; }
            C[(size_t)(mbase + r) * N + nc0]      = v0;
            C[(size_t)(mbase + r) * N + nc0 + 16] = v1;
        }
    } else {
        #pragma unroll
        for (int t = 0; t < 2; ++t) {
            const int ncol = nc0 + t * 16;
            if (ncol < N) {
                const float bv = bias[ncol];
                #pragma unroll
                for (int r = 0; r < 8; ++r) {
                    if (mbase + r < M) {
                        float v = (t == 0 ? acc0[r] : acc1[r]) + bv;
                        if (RELU) v = v > 0.0f ? v : 0.0f;
                        C[(size_t)(mbase + r) * N + ncol] = v;
                    }
                }
            }
        }
    }
}

// ---------------------------------------------------------------------------
// rel_preds: per edge, pair(8) -> hidden(54) -> out(100)
// ---------------------------------------------------------------------------
__global__ void relpred_kernel(const float* __restrict__ boxes,
                               const int* __restrict__ edges,
                               const float* __restrict__ W1, const float* __restrict__ b1,
                               const float* __restrict__ W2, const float* __restrict__ b2,
                               float* __restrict__ out, int B, int Nn, int E)
{
    int idx = blockIdx.x * blockDim.x + threadIdx.x;   // b*E + e
    if (idx >= B * E) return;
    int b = idx / E;
    const int* ed = edges + (size_t)idx * 2;
    int s = ed[0], d = ed[1];
    const float* bs = boxes + ((size_t)b * Nn + s) * 4;
    const float* bd = boxes + ((size_t)b * Nn + d) * 4;
    float pair[8];
    #pragma unroll
    for (int i = 0; i < 4; ++i) {
        pair[i]     = bs[i] * (1.0f / 512.0f);
        pair[4 + i] = bd[i] * (1.0f / 512.0f);
    }
    float hid[54];
    for (int h = 0; h < 54; ++h) {
        float a = b1[h];
        #pragma unroll
        for (int i = 0; i < 8; ++i) a += pair[i] * W1[i * 54 + h];
        hid[h] = a;
    }
    float* o = out + (size_t)idx * 100;
    for (int r = 0; r < 100; ++r) {
        float a = b2[r];
        for (int h = 0; h < 54; ++h) a += hid[h] * W2[h * 100 + r];
        o[r] = a;
    }
}

// ---------------------------------------------------------------------------
// GCN helpers
// ---------------------------------------------------------------------------
__global__ void zero_kernel(float* __restrict__ p, size_t n)
{
    size_t i = (size_t)blockIdx.x * blockDim.x + threadIdx.x;
    if (i < n) p[i] = 0.0f;
}

__global__ void deg_count_kernel(const int* __restrict__ edges, float* __restrict__ deg,
                                 int total, int E, int Nn)
{
    int i = blockIdx.x * blockDim.x + threadIdx.x;   // b*E + e
    if (i >= total) return;
    int b = i / E;
    int dst = edges[(size_t)i * 2 + 1];
    atomicAdd(&deg[(size_t)b * Nn + dst], 1.0f);
}

// one block per edge, threads stride the feature dim
__global__ void agg_scatter_kernel(const float* __restrict__ x, const int* __restrict__ edges,
                                   float* __restrict__ agg, int E, int Nn, int D)
{
    int be = blockIdx.x;                 // b*E + e
    int b  = be / E;
    const int* ed = edges + (size_t)be * 2;
    int s = ed[0], d = ed[1];
    const float* xs = x   + ((size_t)b * Nn + s) * D;
    float*       ad = agg + ((size_t)b * Nn + d) * D;
    for (int i = threadIdx.x; i < D; i += blockDim.x)
        atomicAdd(&ad[i], xs[i]);
}

// agg <- x + agg / max(deg,1)   (in place on agg)
__global__ void combine_kernel(const float* __restrict__ x, float* __restrict__ agg,
                               const float* __restrict__ deg, int D, size_t total)
{
    size_t i = (size_t)blockIdx.x * blockDim.x + threadIdx.x;
    if (i >= total) return;
    size_t node = i / (size_t)D;         // b*Nn + n
    float dg = deg[node];
    if (dg < 1.0f) dg = 1.0f;
    agg[i] = x[i] + agg[i] / dg;
}

// pooled[b,d] = mean_n h[b,n,d] -> fused[b, colOff + d]
__global__ void mean_pool_kernel(const float* __restrict__ h, float* __restrict__ fused,
                                 int B, int Nn, int D, int stride, int colOff)
{
    int i = blockIdx.x * blockDim.x + threadIdx.x;
    if (i >= B * D) return;
    int b = i / D, d = i % D;
    const float* p = h + (size_t)b * Nn * D + d;
    float s = 0.0f;
    for (int n = 0; n < Nn; ++n) s += p[(size_t)n * D];
    fused[(size_t)b * stride + colOff + d] = s / (float)Nn;
}

// ---------------------------------------------------------------------------
// log_softmax: one block per row
// ---------------------------------------------------------------------------
__global__ __launch_bounds__(256)
void log_softmax_kernel(const float* __restrict__ logits, float* __restrict__ out, int N)
{
    __shared__ float red[256];
    const int row = blockIdx.x;
    const float* x = logits + (size_t)row * N;
    float* o = out + (size_t)row * N;

    float mx = -INFINITY;
    for (int i = threadIdx.x; i < N; i += 256) mx = fmaxf(mx, x[i]);
    red[threadIdx.x] = mx; __syncthreads();
    for (int s = 128; s > 0; s >>= 1) {
        if (threadIdx.x < s) red[threadIdx.x] = fmaxf(red[threadIdx.x], red[threadIdx.x + s]);
        __syncthreads();
    }
    mx = red[0]; __syncthreads();

    float sum = 0.0f;
    for (int i = threadIdx.x; i < N; i += 256) sum += expf(x[i] - mx);
    red[threadIdx.x] = sum; __syncthreads();
    for (int s = 128; s > 0; s >>= 1) {
        if (threadIdx.x < s) red[threadIdx.x] += red[threadIdx.x + s];
        __syncthreads();
    }
    const float lse = logf(red[0]) + mx;
    for (int i = threadIdx.x; i < N; i += 256) o[i] = x[i] - lse;
}

// ---------------------------------------------------------------------------
// Host-side helpers
// ---------------------------------------------------------------------------
static inline void run_gemm(hipStream_t st, const float* A, const float* W, const float* bias,
                            float* C, int M, int N, int K, bool relu)
{
    dim3 grid((N + BLK_N - 1) / BLK_N, (M + BLK_M - 1) / BLK_M);
    const bool aligned = (M % BLK_M == 0) && (N % BLK_N == 0) && (K % 4 == 0);
    if (aligned) {
        if (relu) gemm_bias_kernel<true , true ><<<grid, 256, 0, st>>>(A, W, bias, C, M, N, K);
        else      gemm_bias_kernel<false, true ><<<grid, 256, 0, st>>>(A, W, bias, C, M, N, K);
    } else {
        if (relu) gemm_bias_kernel<true , false><<<grid, 256, 0, st>>>(A, W, bias, C, M, N, K);
        else      gemm_bias_kernel<false, false><<<grid, 256, 0, st>>>(A, W, bias, C, M, N, K);
    }
}

static inline void run_zero(hipStream_t st, float* p, size_t n)
{
    zero_kernel<<<(unsigned)((n + 255) / 256), 256, 0, st>>>(p, n);
}

static void run_gcn(hipStream_t st, const float* x, const int* edges,
                    int B, int Nn, int E, int D, int D1, int D2,
                    const float* W1, const float* b1,
                    const float* W2, const float* b2,
                    float* fused, int fstride, int colOff,
                    float* deg, float* S1, float* S2, float* S3, float* S4)
{
    const size_t nd = (size_t)B * Nn;
    run_zero(st, deg, nd);
    deg_count_kernel<<<(B * E + 255) / 256, 256, 0, st>>>(edges, deg, B * E, E, Nn);

    const size_t xtot = nd * (size_t)D;
    run_zero(st, S1, xtot);
    agg_scatter_kernel<<<B * E, 256, 0, st>>>(x, edges, S1, E, Nn, D);
    combine_kernel<<<(unsigned)((xtot + 255) / 256), 256, 0, st>>>(x, S1, deg, D, xtot);
    run_gemm(st, S1, W1, b1, S2, B * Nn, D1, D, /*relu=*/true);

    const size_t h1tot = nd * (size_t)D1;
    run_zero(st, S3, h1tot);
    agg_scatter_kernel<<<B * E, 256, 0, st>>>(S2, edges, S3, E, Nn, D1);
    combine_kernel<<<(unsigned)((h1tot + 255) / 256), 256, 0, st>>>(S2, S3, deg, D1, h1tot);
    run_gemm(st, S3, W2, b2, S4, B * Nn, D2, D1, /*relu=*/false);

    mean_pool_kernel<<<(B * D2 + 255) / 256, 256, 0, st>>>(S4, fused, B, Nn, D2, fstride, colOff);
}

// ---------------------------------------------------------------------------
// Entry point
// ---------------------------------------------------------------------------
extern "C" void kernel_launch(void* const* d_in, const int* in_sizes, int n_in,
                              void* d_out, int out_size, void* d_ws, size_t ws_size,
                              hipStream_t stream)
{
    (void)in_sizes; (void)n_in; (void)out_size; (void)ws_size;

    const float* word_emb   = (const float*)d_in[0];   // [128,30,300]
    const float* rcnn_feats = (const float*)d_in[1];   // [128,100,2048]
    const float* rcnn_boxes = (const float*)d_in[2];   // [128,100,4]
    const int*   syn_edges  = (const int*)  d_in[3];   // [128,64,2]
    const int*   sem_edges  = (const int*)  d_in[4];   // [128,64,2]
    const int*   rel_edges  = (const int*)  d_in[5];   // [128,400,2]
    const float* lstm_W = (const float*)d_in[6];  const float* lstm_b = (const float*)d_in[7];
    const float* syn_W1 = (const float*)d_in[8];  const float* syn_b1 = (const float*)d_in[9];
    const float* syn_W2 = (const float*)d_in[10]; const float* syn_b2 = (const float*)d_in[11];
    const float* sem_W1 = (const float*)d_in[12]; const float* sem_b1 = (const float*)d_in[13];
    const float* sem_W2 = (const float*)d_in[14]; const float* sem_b2 = (const float*)d_in[15];
    const float* feat_W1= (const float*)d_in[16]; const float* feat_b1= (const float*)d_in[17];
    const float* feat_W2= (const float*)d_in[18]; const float* feat_b2= (const float*)d_in[19];
    const float* pos_W1 = (const float*)d_in[20]; const float* pos_b1 = (const float*)d_in[21];
    const float* pos_W2 = (const float*)d_in[22]; const float* pos_b2 = (const float*)d_in[23];
    const float* cls_W1 = (const float*)d_in[24]; const float* cls_b1 = (const float*)d_in[25];
    const float* cls_W2 = (const float*)d_in[26]; const float* cls_b2 = (const float*)d_in[27];
    const float* fus_W1 = (const float*)d_in[28]; const float* fus_b1 = (const float*)d_in[29];
    const float* fus_W2 = (const float*)d_in[30]; const float* fus_b2 = (const float*)d_in[31];

    constexpr int B = 128, T = 30, Nn = 100, E = 400, Et = 64, Es = 64;
    constexpr int Dw = 300, Drc = 2048, R = 100, A = 1842, F = 2344;

    float* ws = (float*)d_ws;
    size_t off = 0;
    auto alloc = [&](size_t n) { float* p = ws + off; off += (n + 63) & ~(size_t)63; return p; };
    float* S1        = alloc((size_t)B * 100 * 2048);  // layer-1 agg/hin (max)
    float* S2        = alloc((size_t)B * 400 * 256);   // h1 (max)
    float* S3        = alloc((size_t)B * 400 * 256);   // layer-2 agg/hin (max)
    float* S4        = alloc((size_t)B * 400 * 256);   // h2 (max)
    float* deg       = alloc((size_t)B * 400);
    float* rel_preds = alloc((size_t)B * E * R);
    float* sem_feats = alloc((size_t)B * T * 512);
    float* fused     = alloc((size_t)B * F);
    float* fc1       = alloc((size_t)B * A);
    float* logits    = alloc((size_t)B * A);

    // 1) rel_preds = (pair @ cls_W1 + b1) @ cls_W2 + b2        [B,400,100]
    relpred_kernel<<<(B * E + 255) / 256, 256, 0, stream>>>(
        rcnn_boxes, rel_edges, cls_W1, cls_b1, cls_W2, cls_b2, rel_preds, B, Nn, E);

    // 2) sem_feats = word_emb @ lstm_W + lstm_b                [B*30, 512]
    run_gemm(stream, word_emb, lstm_W, lstm_b, sem_feats, B * T, 512, Dw, false);

    // 3) four GCNs, pooled directly into the fused buffer
    run_gcn(stream, word_emb,  syn_edges, B, T,   Et, Dw,  512,  512,
            syn_W1,  syn_b1,  syn_W2,  syn_b2,  fused, F, 0,    deg, S1, S2, S3, S4);
    run_gcn(stream, rel_preds, rel_edges, B, 400, E,  R,   256,  256,
            pos_W1,  pos_b1,  pos_W2,  pos_b2,  fused, F, 512,  deg, S1, S2, S3, S4);
    run_gcn(stream, rcnn_feats, rel_edges, B, Nn, E,  Drc, 1024, 1024,
            feat_W1, feat_b1, feat_W2, feat_b2, fused, F, 768,  deg, S1, S2, S3, S4);
    run_gcn(stream, sem_feats, sem_edges, B, T,   Es, 512, 512,  512,
            sem_W1,  sem_b1,  sem_W2,  sem_b2,  fused, F, 1792, deg, S1, S2, S3, S4);

    // 4) fusion MLP + log_softmax
    run_gemm(stream, fused, fus_W1, fus_b1, fc1,    B, A, F, false);
    run_gemm(stream, fc1,   fus_W2, fus_b2, logits, B, A, A, false);
    log_softmax_kernel<<<B, 256, 0, stream>>>(logits, (float*)d_out, A);
}
